// SlidingWindowAttention_55722905699073
// MI455X (gfx1250) — matrix operand order repaired and checked by reading, not compiled, not run on hardware
//
#include <hip/hip_runtime.h>
#include <hip/hip_bf16.h>

#define D_MODEL 2048
#define NUM_HEADS 16
#define HEAD_DIM 128
#define WINDOW 512
#define SINKS 4
#define BATCH 2
#define SEQ 2048
#define MROWS (BATCH * SEQ)

#ifndef USE_ASYNC_LDS
#define USE_ASYNC_LDS 1          // flip to 0 if the async mnemonic fails to assemble
#endif

typedef __attribute__((ext_vector_type(16))) _Float16 v16h;
typedef __attribute__((ext_vector_type(8)))  _Float16 v8h;
typedef __attribute__((ext_vector_type(8)))  float    v8f;
typedef __attribute__((ext_vector_type(4)))  float    v4f;

// ---------------------------------------------------------------------------
// CDNA5 wave32 WMMA layouts (CDNA5 ISA):
//  A (16x32 f16): lane L holds row M=L&15; halves e: k = e + 8*(L>>4)   (e<8)
//                 and k = 16 + (e-8) + 8*(L>>4)                          (e>=8)
//  B (32x16 f16): lane L holds col N=L&15; same k order as A (this code loads A and B with k = e + 16*(L>>4))
//  C/D (16x16 f32): vgpr r, lane L -> (M = r + 8*(L>>4), N = L&15)
// ---------------------------------------------------------------------------

static __device__ inline v16h pack16(v8h lo, v8h hi) {
  v16h a;
#pragma unroll
  for (int e = 0; e < 8; ++e) { a[e] = lo[e]; a[e + 8] = hi[e]; }
  return a;
}

// p = row_base + k0 + 16*(lane>>4); 16 contiguous halves
static __device__ inline v16h load_a_f16(const _Float16* p) {
  return pack16(*(const v8h*)p, *(const v8h*)(p + 8));
}

// p = col_row_base + k0 + 16*(lane>>4); 16 contiguous halves
static __device__ inline v16h load_b_f16(const _Float16* p) {
  return pack16(*(const v8h*)p, *(const v8h*)(p + 8));
}

#define WMMA_F16(A, B, C) \
  __builtin_amdgcn_wmma_f32_16x16x32_f16(false, (A), false, (B), (short)0, (C), false, false)

// ---------------------------------------------------------------------------
// 64B global -> LDS copy per thread.  Async path: 4x
// global_load_async_to_lds_b128 (ASYNCcnt, bypasses VGPRs; IOFFSET applies to
// both the global and the LDS address per CDNA5 ISA).
// ---------------------------------------------------------------------------
static __device__ inline void stage64(const _Float16* g, _Float16* l) {
#if USE_ASYNC_LDS
  const unsigned int       la = (unsigned int)(unsigned long long)l;   // low 32 bits = LDS offset
  const unsigned long long ga = (unsigned long long)g;
  asm volatile(
      "global_load_async_to_lds_b128 %0, %1, off\n\t"
      "global_load_async_to_lds_b128 %0, %1, off offset:16\n\t"
      "global_load_async_to_lds_b128 %0, %1, off offset:32\n\t"
      "global_load_async_to_lds_b128 %0, %1, off offset:48"
      :: "v"(la), "v"(ga) : "memory");
#else
  const uint4* gs = (const uint4*)g;
  uint4* ls = (uint4*)l;
  uint4 a = gs[0], b = gs[1], c = gs[2], d = gs[3];
  ls[0] = a; ls[1] = b; ls[2] = c; ls[3] = d;
#endif
}

static __device__ inline void tile_barrier() {
#if USE_ASYNC_LDS
  asm volatile("s_wait_asynccnt 0" ::: "memory");
#endif
  __syncthreads();
}

// ---------------------------------------------------------------------------
// Kernel 0: elementwise f32 -> f16 (one-time).
// ---------------------------------------------------------------------------
__global__ __launch_bounds__(256) void cvt_f16_kernel(
    const float* __restrict__ in, _Float16* __restrict__ out)
{
  const size_t i = ((size_t)blockIdx.x * 256 + threadIdx.x) * 8;
  v4f x0 = *(const v4f*)(in + i);
  v4f x1 = *(const v4f*)(in + i + 4);
  v8h o;
#pragma unroll
  for (int e = 0; e < 4; ++e) { o[e] = (_Float16)x0[e]; o[e + 4] = (_Float16)x1[e]; }
  *(v8h*)(out + i) = o;
}

// ---------------------------------------------------------------------------
// Shared f16 GEMM core: one wave computes a 32x64 tile.  Distinct fragment
// registers -> loads clause and overlap (verified in round-2 asm).
// ---------------------------------------------------------------------------
static __device__ inline void gemm32x64_f16(
    const _Float16* __restrict__ A, const _Float16* __restrict__ W,
    int m0, int n0, int lane, v8f acc[8])
{
  const int hi = lane >> 4, nl = lane & 15;
  const _Float16* arow0 = A + (size_t)(m0 + nl) * D_MODEL + 16 * hi;
  const _Float16* arow1 = arow0 + (size_t)16 * D_MODEL;
  const _Float16* wrow  = W + (size_t)(n0 + nl) * D_MODEL + 16 * hi;

  for (int k0 = 0; k0 < D_MODEL; k0 += 32) {
    v16h a0 = load_a_f16(arow0 + k0);
    v16h a1 = load_a_f16(arow1 + k0);
    v16h b0 = load_b_f16(wrow + k0);
    v16h b1 = load_b_f16(wrow + (size_t)16 * D_MODEL + k0);
    v16h b2 = load_b_f16(wrow + (size_t)32 * D_MODEL + k0);
    v16h b3 = load_b_f16(wrow + (size_t)48 * D_MODEL + k0);
    acc[0] = WMMA_F16(a0, b0, acc[0]);
    acc[1] = WMMA_F16(a0, b1, acc[1]);
    acc[2] = WMMA_F16(a0, b2, acc[2]);
    acc[3] = WMMA_F16(a0, b3, acc[3]);
    acc[4] = WMMA_F16(a1, b0, acc[4]);
    acc[5] = WMMA_F16(a1, b1, acc[5]);
    acc[6] = WMMA_F16(a1, b2, acc[6]);
    acc[7] = WMMA_F16(a1, b3, acc[7]);
  }
}

// ---------------------------------------------------------------------------
// Kernel 1: y = Xh @ Wh^T + bias (f16 out), optional V-transpose layout.
// grid (MROWS/128, D_MODEL/64), block 128.
// ---------------------------------------------------------------------------
__global__ __launch_bounds__(128) void qkv_proj_kernel(
    const _Float16* __restrict__ Xh, const _Float16* __restrict__ Wh,
    const float* __restrict__ bias, _Float16* __restrict__ out,
    int v_transposed, float out_scale)
{
  const int lane = threadIdx.x & 31;
  const int wave = threadIdx.x >> 5;
  const int m0 = blockIdx.x * 128 + wave * 32;
  const int n0 = blockIdx.y * 64;
  const int hi = lane >> 4, nl = lane & 15;

  v8f acc[8] = {};
  gemm32x64_f16(Xh, Wh, m0, n0, lane, acc);

#pragma unroll
  for (int t = 0; t < 4; ++t) {
    const int n = n0 + 16 * t + nl;
    const float bn = bias[n];
    const int h = n >> 7, d = n & 127;
#pragma unroll
    for (int mt = 0; mt < 2; ++mt) {
#pragma unroll
      for (int r = 0; r < 8; ++r) {
        const int m = m0 + mt * 16 + r + 8 * hi;
        const int b = m >> 11, s = m & 2047;
        const float v = (acc[mt * 4 + t][r] + bn) * out_scale;
        const size_t idx = v_transposed
            ? ((size_t)((b * NUM_HEADS + h) * HEAD_DIM + d)) * SEQ + s
            : ((size_t)((b * NUM_HEADS + h) * SEQ + s)) * HEAD_DIM + d;
        out[idx] = (_Float16)v;
      }
    }
  }
}

// ---------------------------------------------------------------------------
// Kernel 2: sliding-window-with-sinks attention (flash style, f16 WMMA).
// grid (SEQ/64, BATCH*NUM_HEADS), block 128; each wave owns a 16-row q tile.
// K/V chunks staged once per block into double-buffered LDS (async loads),
// shared by all 4 waves -> 4x less L2 traffic.
// ---------------------------------------------------------------------------
#define KPITCH 136   // 32x128 K tile, padded rows (272B, 16B-aligned, bank-spread)
#define VPITCH 40    // 128x32 V^T tile, padded rows (80B)

__global__ __launch_bounds__(128) void swa_attn_kernel(
    const _Float16* __restrict__ qh, const _Float16* __restrict__ kh,
    const _Float16* __restrict__ vt, _Float16* __restrict__ attn)
{
  __shared__ alignas(16) _Float16 Kbuf[2][32 * KPITCH];
  __shared__ alignas(16) _Float16 Vbuf[2][128 * VPITCH];
  __shared__ alignas(16) _Float16 Pbuf[4][16 * 48];

  const int tid  = threadIdx.x;
  const int lane = tid & 31;
  const int wave = tid >> 5;
  const int hi = lane >> 4, nl = lane & 15;
  const int bh = blockIdx.y;
  const int b = bh >> 4, h = bh & 15;
  const int q0blk = blockIdx.x * 64;
  const int q0 = q0blk + wave * 16;

  const _Float16* Q = qh + (size_t)bh * SEQ * HEAD_DIM;
  const _Float16* K = kh + (size_t)bh * SEQ * HEAD_DIM;
  const _Float16* V = vt + (size_t)bh * HEAD_DIM * SEQ;   // [d][s]
  _Float16* lds = &Pbuf[wave][0];

  // Block-uniform chunk schedule: chunk 0 (sinks) + chunks >= c1.
  // Per-element masking below handles rows whose window is stricter.
  int c1 = (q0blk - WINDOW) >> 5;
  if (c1 < 1) c1 = 1;
  const int nchunks = 1 + (SEQ / 32 - c1);

  // staging: K tile row = tid/4, 32-half quarter = tid%4 ; V tile row d = tid
  const int krow = tid >> 2, kq = tid & 3;

  // Q fragments (1/sqrt(d) folded in at projection)
  v16h qf[4];
#pragma unroll
  for (int kc = 0; kc < 4; ++kc)
    qf[kc] = load_a_f16(Q + (size_t)(q0 + nl) * HEAD_DIM + kc * 32 + 16 * hi);

  v8f acc[8] = {};                 // O tile 16x128 (8 d-subtiles)
  float mrun[8], lrun[8];
#pragma unroll
  for (int r = 0; r < 8; ++r) { mrun[r] = -3.0e38f; lrun[r] = 0.0f; }

  // prefetch chunk 0 into buffer 0
  stage64(K + (size_t)krow * HEAD_DIM + kq * 32, &Kbuf[0][krow * KPITCH + kq * 32]);
  stage64(V + (size_t)tid * SEQ, &Vbuf[0][tid * VPITCH]);

  for (int idx = 0; idx < nchunks; ++idx) {
    const int cur = idx & 1;
    const int kv0 = (idx == 0) ? 0 : (c1 + idx - 1) * 32;

    tile_barrier();   // buf[cur] filled; all waves done reading buf[cur^1]'s old data

    if (idx + 1 < nchunks) {
      const int kvn = (c1 + idx) * 32;
      stage64(K + (size_t)(kvn + krow) * HEAD_DIM + kq * 32,
              &Kbuf[cur ^ 1][krow * KPITCH + kq * 32]);
      stage64(V + (size_t)tid * SEQ + kvn, &Vbuf[cur ^ 1][tid * VPITCH]);
    }

    const _Float16* Kl = &Kbuf[cur][0];
    const _Float16* Vl = &Vbuf[cur][0];

    // S tile 16x32 = Q(16x128) @ K^T(128x32), K from LDS
    v8f s0 = {}, s1 = {};
#pragma unroll
    for (int kc = 0; kc < 4; ++kc) {
      v16h b0 = load_b_f16(Kl + nl * KPITCH + kc * 32 + 16 * hi);
      v16h b1 = load_b_f16(Kl + (16 + nl) * KPITCH + kc * 32 + 16 * hi);
      s0 = WMMA_F16(qf[kc], b0, s0);
      s1 = WMMA_F16(qf[kc], b1, s1);
    }

    const int j0 = kv0 + nl, j1 = kv0 + 16 + nl;
#pragma unroll
    for (int r = 0; r < 8; ++r) {
      const int m = q0 + r + 8 * hi;
      const int lim = m - WINDOW;
      // banned iff j >= SINKS && j < pos - WINDOW   (reference mask, non-causal)
      if (j0 >= SINKS && j0 < lim) s0[r] = -3.0e38f;
      if (j1 >= SINKS && j1 < lim) s1[r] = -3.0e38f;

      float mx = fmaxf(s0[r], s1[r]);
      mx = fmaxf(mx, __shfl_xor(mx, 1));
      mx = fmaxf(mx, __shfl_xor(mx, 2));
      mx = fmaxf(mx, __shfl_xor(mx, 4));
      mx = fmaxf(mx, __shfl_xor(mx, 8));     // stays inside 16-lane row group

      const float mnew = fmaxf(mrun[r], mx); // finite: chunk 0 (sinks) first
      const float corr = __expf(mrun[r] - mnew);
      mrun[r] = mnew;

      const float p0 = __expf(s0[r] - mnew);
      const float p1 = __expf(s1[r] - mnew);
      float rs = p0 + p1;
      rs += __shfl_xor(rs, 1);
      rs += __shfl_xor(rs, 2);
      rs += __shfl_xor(rs, 4);
      rs += __shfl_xor(rs, 8);
      lrun[r] = lrun[r] * corr + rs;

#pragma unroll
      for (int t = 0; t < 8; ++t) acc[t][r] *= corr;

      // C-layout -> LDS (row-major 16x32 P tile, private per wave)
      lds[(r + 8 * hi) * 48 + nl]      = (_Float16)p0;
      lds[(r + 8 * hi) * 48 + 16 + nl] = (_Float16)p1;
    }

    asm volatile("s_wait_dscnt 0" ::: "memory");   // same-wave LDS cross-lane RAW

    // Reload P as A-fragment (16x32), then O += P @ V (V from LDS)
    v16h pf = load_a_f16(lds + nl * 48 + 16 * hi);
#pragma unroll
    for (int t = 0; t < 8; ++t) {
      v16h bv = load_b_f16(Vl + (t * 16 + nl) * VPITCH + 16 * hi);
      acc[t] = WMMA_F16(pf, bv, acc[t]);
    }
  }

  // normalize and store f16 activations, row-major [b*SEQ+s][h*128+d]
#pragma unroll
  for (int r = 0; r < 8; ++r) {
    const float inv = 1.0f / lrun[r];
    const int s = q0 + r + 8 * hi;
    _Float16* orow = attn + ((size_t)(b * SEQ + s)) * D_MODEL + h * HEAD_DIM;
#pragma unroll
    for (int t = 0; t < 8; ++t)
      orow[t * 16 + nl] = (_Float16)(acc[t][r] * inv);
  }
}

// ---------------------------------------------------------------------------
// Kernel 3: out = attn(f16) @ Woh^T + bo   (f32 output)
// ---------------------------------------------------------------------------
__global__ __launch_bounds__(128) void out_proj_kernel(
    const _Float16* __restrict__ A, const _Float16* __restrict__ Wh,
    const float* __restrict__ bias, float* __restrict__ out)
{
  const int lane = threadIdx.x & 31;
  const int wave = threadIdx.x >> 5;
  const int m0 = blockIdx.x * 128 + wave * 32;
  const int n0 = blockIdx.y * 64;
  const int hi = lane >> 4, nl = lane & 15;

  v8f acc[8] = {};
  gemm32x64_f16(A, Wh, m0, n0, lane, acc);

#pragma unroll
  for (int t = 0; t < 4; ++t) {
    const int n = n0 + 16 * t + nl;
    const float bn = bias[n];
#pragma unroll
    for (int mt = 0; mt < 2; ++mt) {
#pragma unroll
      for (int r = 0; r < 8; ++r) {
        const int m = m0 + mt * 16 + r + 8 * hi;
        out[(size_t)m * D_MODEL + n] = acc[mt * 4 + t][r] + bn;
      }
    }
  }
}

// ---------------------------------------------------------------------------
extern "C" void kernel_launch(void* const* d_in, const int* in_sizes, int n_in,
                              void* d_out, int out_size, void* d_ws, size_t ws_size,
                              hipStream_t stream) {
  const float* X  = (const float*)d_in[0];
  const float* Wq = (const float*)d_in[1];
  const float* bq = (const float*)d_in[2];
  const float* Wk = (const float*)d_in[3];
  const float* bk = (const float*)d_in[4];
  const float* Wv = (const float*)d_in[5];
  const float* bv = (const float*)d_in[6];
  const float* Wo = (const float*)d_in[7];
  const float* bo = (const float*)d_in[8];
  float* out = (float*)d_out;

  const size_t act = (size_t)BATCH * SEQ * D_MODEL;   // 8M elements
  const size_t wgt = (size_t)D_MODEL * D_MODEL;       // 4M elements
  _Float16* qh   = (_Float16*)d_ws;
  _Float16* kh   = qh + act;
  _Float16* vt   = kh + act;
  _Float16* attn = vt + act;
  _Float16* Xh   = attn + act;
  _Float16* Wqh  = Xh + act;
  _Float16* Wkh  = Wqh + wgt;
  _Float16* Wvh  = Wkh + wgt;
  _Float16* Woh  = Wvh + wgt;

  const dim3 blk(128);

  cvt_f16_kernel<<<dim3(act / (256 * 8)), dim3(256), 0, stream>>>(X, Xh);
  cvt_f16_kernel<<<dim3(wgt / (256 * 8)), dim3(256), 0, stream>>>(Wq, Wqh);
  cvt_f16_kernel<<<dim3(wgt / (256 * 8)), dim3(256), 0, stream>>>(Wk, Wkh);
  cvt_f16_kernel<<<dim3(wgt / (256 * 8)), dim3(256), 0, stream>>>(Wv, Wvh);
  cvt_f16_kernel<<<dim3(wgt / (256 * 8)), dim3(256), 0, stream>>>(Wo, Woh);

  const dim3 gL(MROWS / 128, D_MODEL / 64);
  const float scale = 0.08838834764831845f;  // HEAD_DIM^-0.5, folded into Q

  qkv_proj_kernel<<<gL, blk, 0, stream>>>(Xh, Wqh, bq, qh, 0, scale);
  qkv_proj_kernel<<<gL, blk, 0, stream>>>(Xh, Wkh, bk, kh, 0, 1.0f);
  qkv_proj_kernel<<<gL, blk, 0, stream>>>(Xh, Wvh, bv, vt, 1, 1.0f);

  const dim3 gA(SEQ / 64, BATCH * NUM_HEADS);
  swa_attn_kernel<<<gA, blk, 0, stream>>>(qh, kh, vt, attn);

  out_proj_kernel<<<gL, blk, 0, stream>>>(attn, Woh, bo, out);
}
